// Eagle3Attention_24111946399989
// MI455X (gfx1250) — matrix operand order repaired and checked by reading, not compiled, not run on hardware
//
#include <hip/hip_runtime.h>
#include <hip/hip_bf16.h>

// ---------------- types ----------------
typedef __attribute__((ext_vector_type(16))) __bf16 v16bf;
typedef __attribute__((ext_vector_type(8)))  __bf16 v8bf;
typedef __attribute__((ext_vector_type(8)))  float  v8f;

union V16 { v16bf v; v8bf h[2]; };

#if defined(__has_builtin)
#if __has_builtin(__builtin_amdgcn_global_load_async_to_lds_b128) && \
    __has_builtin(__builtin_amdgcn_s_wait_asynccnt)
#define GEMM_ASYNC 1
#endif
#endif

#ifdef GEMM_ASYNC
// Builtin signature (leaked by the round-2 diagnostic): param 1 is
// 'int __attribute__((vector_size(16))) __device__ *' (global AS1); LDS side is AS3.
typedef int v4i __attribute__((vector_size(16)));
typedef __attribute__((address_space(1))) v4i gl_v4i;
typedef __attribute__((address_space(3))) v4i lds_v4i;

static __device__ __forceinline__ void async_b128(const __bf16* g, __bf16* l) {
  __builtin_amdgcn_global_load_async_to_lds_b128((gl_v4i*)(void*)g,
                                                 (lds_v4i*)(void*)l, 0, 0);
}
#endif

// ---------------- problem constants ----------------
enum : int {
  LQ  = 2048,   // sequence length
  CIN = 4096,   // input features (2*HID)
  NH  = 32,     // query heads
  NKV = 8,      // kv heads
  HD  = 64,     // head dim
  HID = 2048,   // hidden size (output)
};

static __device__ __forceinline__ v8f wmma_bf16(v16bf a, v16bf b, v8f c) {
  return __builtin_amdgcn_wmma_f32_16x16x32_bf16(false, a, false, b, (short)0, c,
                                                 false, false);
}

// ---------------- fp32 -> bf16 elementwise ----------------
__global__ __launch_bounds__(256) void cvt_bf16_kernel(const float* __restrict__ in,
                                                       __bf16* __restrict__ out, int n) {
  for (int i = blockIdx.x * blockDim.x + threadIdx.x; i < n; i += gridDim.x * blockDim.x)
    out[i] = (__bf16)in[i];
}

// ---------------- fp32 [K x N] -> bf16 transposed [N x K] (LDS tile) -------------
__global__ __launch_bounds__(256) void cvt_t_bf16_kernel(const float* __restrict__ in,
                                                         __bf16* __restrict__ out,
                                                         int K, int N) {
  __shared__ float tile[32][33];
  int kb = blockIdx.x * 32, nb = blockIdx.y * 32;
  int tx = threadIdx.x, ty = threadIdx.y;        // block (32, 8)
  for (int i = ty; i < 32; i += 8)
    tile[i][tx] = in[(size_t)(kb + i) * N + (nb + tx)];
  __syncthreads();
  for (int i = ty; i < 32; i += 8)
    out[(size_t)(nb + i) * K + (kb + tx)] = (__bf16)tile[tx][i];
}

// ---------------- bf16 GEMM: C[M,N] = A[M,K] * Bt[N,K]^T ----------------
// Block: 256 threads = 8 waves. Tile: M=128 (16 rows/wave), N=128 (8 wmma accs/wave),
// K-step 32. B tile staged in LDS via GLOBAL_LOAD_ASYNC_TO_LDS, double-buffered,
// one barrier + one s_wait_asynccnt per k-step; copy overlaps compute.
// mode 0: bf16 out, head-major  out[(n>>6)*(M*64) + m*64 + (n&63)]
// mode 1: bf16 out, transposed  out[(n>>6)*(64*M) + (n&63)*M + m]   (for V^T)
// mode 2: f32  out, row-major   out[m*N + n]
__global__ __launch_bounds__(256) void gemm_bf16_kernel(const __bf16* __restrict__ A,
                                                        const __bf16* __restrict__ Bt,
                                                        void* __restrict__ Cout,
                                                        int M, int N, int K, int mode) {
#ifdef GEMM_ASYNC
  __shared__ __attribute__((aligned(16))) __bf16 lB[2][128][48];  // double buffer
#else
  __shared__ __attribute__((aligned(16))) __bf16 lB[1][128][48];
#endif

  const int tid  = threadIdx.x;
  const int lane = tid & 31;
  const int hl   = lane >> 4;       // half-wave (0/1)
  const int nl   = lane & 15;       // lane in half
  const int m0   = blockIdx.x * 128 + (tid >> 5) * 16;
  const int n0   = blockIdx.y * 128;

  v8f acc[8] = {};
  const __bf16* arow = A + (size_t)(m0 + nl) * K + hl * 16;

#ifdef GEMM_ASYNC
  // Each thread owns two fixed 16-byte chunks of the 128x32 B tile (8 KB total).
  const int r0 = tid >> 2,          o0 = (tid & 3) * 8;
  const int r1 = (tid + 256) >> 2,  o1 = (tid & 3) * 8;   // same k-offset, rows 64..127
  const __bf16* gB0 = Bt + (size_t)(n0 + r0) * K + o0;    // advance by k0 only
  const __bf16* gB1 = Bt + (size_t)(n0 + r1) * K + o1;
  __bf16* lD[2][2] = {{&lB[0][r0][o0], &lB[0][r1][o1]},
                      {&lB[1][r0][o0], &lB[1][r1][o1]}};

  // prologue: tile k=0 into buffer 0
  async_b128(gB0, lD[0][0]);
  async_b128(gB1, lD[0][1]);

  auto step = [&](int k0, int buf) {
    __builtin_amdgcn_s_wait_asynccnt(0);  // this wave's chunks of lB[buf] landed
    __syncthreads();                      // all waves' chunks landed; prev buf free
    // prefetch next tile (wrapped to k=0 on the last step: in-bounds, lands in the
    // buffer that is never read again; kernel-end wait-idle covers it)
    int kn = (k0 + 32 < K) ? (k0 + 32) : 0;
    async_b128(gB0 + kn, lD[buf ^ 1][0]);
    async_b128(gB1 + kn, lD[buf ^ 1][1]);

    // A fragment: lane holds row (m0 + nl); k = {hl*16..+16}
    V16 af;
    af.h[0] = *(const v8bf*)(arow + k0);
    af.h[1] = *(const v8bf*)(arow + k0 + 8);

    // batch the 8 B-fragment LDS loads, then 8 WMMAs
    v16bf bfr[8];
    const __bf16* lbase = &lB[buf][nl][hl * 16];
    for (int j = 0; j < 8; ++j)
      bfr[j] = *(const v16bf*)(lbase + j * (16 * 48));
    for (int j = 0; j < 8; ++j)
      acc[j] = wmma_bf16(af.v, bfr[j], acc[j]);
  };

  for (int k0 = 0; k0 < K; k0 += 64) {   // K is a multiple of 64
    step(k0, 0);
    step(k0 + 32, 1);
  }
#else
  for (int k0 = 0; k0 < K; k0 += 32) {
    __syncthreads();   // protect lB WAR from previous iteration
    for (int c = tid; c < 512; c += 256) {
      int row = c >> 2, ko = (c & 3) * 8;
      const __bf16* src = Bt + (size_t)(n0 + row) * K + k0 + ko;
      *(v8bf*)&lB[0][row][ko] = *(const v8bf*)src;
      if (k0 + 32 < K) __builtin_prefetch(src + 32, 0, 3);
    }
    __syncthreads();

    V16 af;
    af.h[0] = *(const v8bf*)(arow + k0);
    af.h[1] = *(const v8bf*)(arow + k0 + 8);

    for (int j = 0; j < 8; ++j) {
      v16bf bfr = *(const v16bf*)&lB[0][16 * j + nl][hl * 16];
      acc[j] = wmma_bf16(af.v, bfr, acc[j]);
    }
  }
#endif

  // Writeback. C/D layout: lane -> (m = r + 8*hl, n = nl) for vgpr r.
  for (int j = 0; j < 8; ++j) {
    for (int r = 0; r < 8; ++r) {
      int m  = m0 + r + 8 * hl;
      int nn = n0 + 16 * j + nl;
      float v = acc[j][r];
      if (mode == 2) {
        ((float*)Cout)[(size_t)m * N + nn] = v;
      } else if (mode == 0) {
        ((__bf16*)Cout)[(size_t)(nn >> 6) * ((size_t)M * 64) + (size_t)m * 64 + (nn & 63)] =
            (__bf16)v;
      } else {
        ((__bf16*)Cout)[(size_t)(nn >> 6) * ((size_t)64 * M) + (size_t)(nn & 63) * M + m] =
            (__bf16)v;
      }
    }
  }
}

// ---------------- attention: flash-style, one wave = 16 queries x D=64 of one head ----
// Q: [NH][L][64] bf16, K: [NKV][L][64] bf16, Vt: [NKV][64][L] bf16, AO: [L][NH*64] bf16
__global__ __launch_bounds__(128) void attn_kernel(const __bf16* __restrict__ Q,
                                                   const __bf16* __restrict__ Kb,
                                                   const __bf16* __restrict__ Vt,
                                                   __bf16* __restrict__ AO) {
  __shared__ __attribute__((aligned(16))) __bf16 lP[4][16][48];  // per-wave 16x32 P tile

  const int tid  = threadIdx.x;
  const int wave = tid >> 5;
  const int lane = tid & 31;
  const int hl   = lane >> 4;
  const int nl   = lane & 15;

  const int wid = blockIdx.x * 4 + wave;       // 4096 waves total
  const int h   = wid >> 7;                    // head (128 q-tiles per head)
  const int q0  = (wid & 127) << 4;            // query tile base
  const int kvh = h >> 2;                      // GQA: repeat_interleave -> h // 4

  const __bf16* Qh = Q  + (size_t)h   * LQ * 64;
  const __bf16* Kh = Kb + (size_t)kvh * LQ * 64;
  const __bf16* Vh = Vt + (size_t)kvh * 64 * LQ;

  // Q fragments for the two K=32 steps across D=64
  V16 qf[2];
  for (int s = 0; s < 2; ++s) {
    const __bf16* qp = Qh + (size_t)(q0 + nl) * 64 + 32 * s + hl * 16;
    qf[s].h[0] = *(const v8bf*)qp;
    qf[s].h[1] = *(const v8bf*)(qp + 8);
  }

  v8f o[4] = {};
  float mrow[8], lrow[8];
  for (int r = 0; r < 8; ++r) { mrow[r] = -1e30f; lrow[r] = 0.0f; }
  const float scale = 0.125f;  // 1/sqrt(64)

  for (int kk = 0; kk < LQ; kk += 32) {
    // ---- S = Q * K^T for 32 keys (two 16x16 tiles) ----
    v8f s0 = {}, s1 = {};
    for (int s = 0; s < 2; ++s) {
      v16bf k0f = *(const v16bf*)(Kh + (size_t)(kk + nl) * 64 + 32 * s + hl * 16);
      v16bf k1f = *(const v16bf*)(Kh + (size_t)(kk + 16 + nl) * 64 + 32 * s + hl * 16);
      s0 = wmma_bf16(qf[s].v, k0f, s0);
      s1 = wmma_bf16(qf[s].v, k1f, s1);
    }

    // ---- fp32 online softmax; rows live per 16-lane half ----
    float corr[8];
    for (int r = 0; r < 8; ++r) {
      float a = s0[r] * scale, b = s1[r] * scale;
      float mx = fmaxf(a, b);
      for (int off = 8; off; off >>= 1) mx = fmaxf(mx, __shfl_xor(mx, off, 32));
      float mn = fmaxf(mrow[r], mx);
      float c  = __expf(mrow[r] - mn);
      float p0 = __expf(a - mn);
      float p1 = __expf(b - mn);
      float sm = p0 + p1;
      for (int off = 8; off; off >>= 1) sm += __shfl_xor(sm, off, 32);
      lrow[r] = lrow[r] * c + sm;
      mrow[r] = mn;
      corr[r] = c;
      // stash P (D-layout -> LDS) for A-layout reload
      lP[wave][r + 8 * hl][nl]      = (__bf16)p0;
      lP[wave][r + 8 * hl][16 + nl] = (__bf16)p1;
    }
    for (int j = 0; j < 4; ++j)
      for (int r = 0; r < 8; ++r) o[j][r] *= corr[r];

    __syncthreads();  // P stores visible (cross-lane) before A-layout reads

    // P as A-fragment: lane row = nl, k(keys) = {hl*16..+16}
    V16 pf;
    pf.h[0] = *(const v8bf*)&lP[wave][nl][hl * 16];
    pf.h[1] = *(const v8bf*)&lP[wave][nl][hl * 16 + 8];

    // ---- O += P * V : B columns are dims d, contiguous keys from Vt ----
    for (int j = 0; j < 4; ++j) {
      v16bf vf = *(const v16bf*)(Vh + (size_t)(16 * j + nl) * LQ + kk + hl * 16);
      o[j] = wmma_bf16(pf.v, vf, o[j]);
    }
    __syncthreads();  // protect lP before next iteration's stores
  }

  // normalize and write attn output [L][NH*64] bf16
  for (int r = 0; r < 8; ++r) {
    float inv = 1.0f / lrow[r];
    int q = q0 + r + 8 * hl;
    for (int j = 0; j < 4; ++j) {
      AO[(size_t)q * (NH * 64) + h * 64 + 16 * j + nl] = (__bf16)(o[j][r] * inv);
    }
  }
}

// ---------------- host side ----------------
extern "C" void kernel_launch(void* const* d_in, const int* in_sizes, int n_in,
                              void* d_out, int out_size, void* d_ws, size_t ws_size,
                              hipStream_t stream) {
  (void)in_sizes; (void)n_in; (void)out_size; (void)ws_size;
  const float* X  = (const float*)d_in[0];   // [L, CIN]
  const float* Wq = (const float*)d_in[1];   // [CIN, NH*HD]
  const float* Wk = (const float*)d_in[2];   // [CIN, NKV*HD]
  const float* Wv = (const float*)d_in[3];   // [CIN, NKV*HD]
  const float* Wo = (const float*)d_in[4];   // [NH*HD, HID]
  float* out = (float*)d_out;                // [L, HID]

  char* ws = (char*)d_ws;
  size_t off = 0;
  auto alloc = [&](size_t bytes) -> void* {
    void* p = ws + off;
    off += (bytes + 255) & ~(size_t)255;
    return p;
  };
  __bf16* Xbf  = (__bf16*)alloc((size_t)LQ * CIN * 2);
  __bf16* Wqt  = (__bf16*)alloc((size_t)(NH * HD) * CIN * 2);
  __bf16* Wkt  = (__bf16*)alloc((size_t)(NKV * HD) * CIN * 2);
  __bf16* Wvt  = (__bf16*)alloc((size_t)(NKV * HD) * CIN * 2);
  __bf16* Wot  = (__bf16*)alloc((size_t)HID * (NH * HD) * 2);
  __bf16* Qb   = (__bf16*)alloc((size_t)NH * LQ * HD * 2);
  __bf16* Kbuf = (__bf16*)alloc((size_t)NKV * LQ * HD * 2);
  __bf16* Vtb  = (__bf16*)alloc((size_t)NKV * HD * LQ * 2);
  __bf16* AO   = (__bf16*)alloc((size_t)LQ * NH * HD * 2);

  // 1) bf16 conversions (+ weight transposes so WMMA B-frags are contiguous)
  cvt_bf16_kernel<<<4096, 256, 0, stream>>>(X, Xbf, LQ * CIN);
  {
    dim3 blk(32, 8);
    cvt_t_bf16_kernel<<<dim3(CIN / 32, (NH * HD) / 32), blk, 0, stream>>>(Wq, Wqt, CIN, NH * HD);
    cvt_t_bf16_kernel<<<dim3(CIN / 32, (NKV * HD) / 32), blk, 0, stream>>>(Wk, Wkt, CIN, NKV * HD);
    cvt_t_bf16_kernel<<<dim3(CIN / 32, (NKV * HD) / 32), blk, 0, stream>>>(Wv, Wvt, CIN, NKV * HD);
    cvt_t_bf16_kernel<<<dim3((NH * HD) / 32, HID / 32), blk, 0, stream>>>(Wo, Wot, NH * HD, HID);
  }

  // 2) projections (bf16 WMMA, fp32 accumulate)
  gemm_bf16_kernel<<<dim3(LQ / 128, (NH * HD) / 128), 256, 0, stream>>>(
      Xbf, Wqt, Qb, LQ, NH * HD, CIN, /*head-major*/ 0);
  gemm_bf16_kernel<<<dim3(LQ / 128, (NKV * HD) / 128), 256, 0, stream>>>(
      Xbf, Wkt, Kbuf, LQ, NKV * HD, CIN, /*head-major*/ 0);
  gemm_bf16_kernel<<<dim3(LQ / 128, (NKV * HD) / 128), 256, 0, stream>>>(
      Xbf, Wvt, Vtb, LQ, NKV * HD, CIN, /*transposed V*/ 1);

  // 3) attention (flash-style online softmax, WMMA for QK^T and P*V)
  attn_kernel<<<dim3((NH * (LQ / 16)) / 4), 128, 0, stream>>>(Qb, Kbuf, Vtb, AO);

  // 4) output projection -> fp32 d_out
  gemm_bf16_kernel<<<dim3(LQ / 128, HID / 128), 256, 0, stream>>>(
      AO, Wot, out, LQ, HID, NH * HD, /*f32 row-major*/ 2);
}
